// MPEncoder_34978213659211
// MI455X (gfx1250) — compile-verified, matrix operand order checked by reading, not run on hardware
//
#include <hip/hip_runtime.h>
#include <hip/hip_bf16.h>

#define DLAT 128
#define TM 16

typedef __attribute__((ext_vector_type(16))) __bf16 v16bf;
typedef __attribute__((ext_vector_type(8)))  __bf16 v8bf;
typedef __attribute__((ext_vector_type(2)))  __bf16 bf16x2;
typedef __attribute__((ext_vector_type(8)))  float  v8f;

__device__ __forceinline__ __bf16 f2bf(float f) {
    unsigned u = __float_as_uint(f);
    u += 0x7fffu + ((u >> 16) & 1u);            // round-to-nearest-even
    unsigned short h = (unsigned short)(u >> 16);
    __bf16 r;
    __builtin_memcpy(&r, &h, 2);
    return r;
}

// ---------------------------------------------------------------------------
// fp32 -> bf16 elementwise (packed: float2 in, 32-bit bf16x2 out)
// ---------------------------------------------------------------------------
__global__ void cvt_f32_bf16(const float* __restrict__ src, __bf16* __restrict__ dst, int n2) {
    int i = blockIdx.x * blockDim.x + threadIdx.x;    // over pairs
    if (i >= n2) return;
    float2 f = ((const float2*)src)[i];
    bf16x2 o;
    o.x = f2bf(f.x);
    o.y = f2bf(f.y);
    ((bf16x2*)dst)[i] = o;
}

// Weight conversion: W[k][n] fp32 -> WbT[n][k] bf16 (transposed: B-fragment
// loads become two contiguous 16B loads per lane)
__global__ void cvt_weight_T(const float* __restrict__ W, __bf16* __restrict__ WbT) {
    int idx = blockIdx.x * blockDim.x + threadIdx.x;  // 128*128
    if (idx >= DLAT * DLAT) return;
    int k = idx >> 7, n = idx & 127;
    WbT[n * DLAT + k] = f2bf(W[k * DLAT + n]);
}

// ---------------------------------------------------------------------------
// Fused GEMM: C[M,128] = act(A[M,128]_bf16 @ W + bias)
// ACT: 0 = identity, 1 = sigmoid, 2 = softplus(x - 5)
// OUTBF: 1 -> bf16 output, 0 -> fp32 output
// Block = 256 threads (8 waves): block -> one 16-row strip (staged in LDS),
// wave -> one 16-col tile, K=128 in 4 WMMA steps.
// 16-bit fragment layout (CDNA5 ISA): lane%16 = M/N, lane/16 = K half,
// fragment = two contiguous 8xbf16 loads at [row, ko+hv*8] and +16.
// ---------------------------------------------------------------------------
template <int ACT, int OUTBF>
__global__ void gemm_bf16_wmma(const __bf16* __restrict__ A,
                               const __bf16* __restrict__ WbT,
                               const float* __restrict__ bias,
                               void* __restrict__ Cout) {
    __shared__ __bf16 sA[TM * DLAT];                  // 4 KB strip
    const int tileM = blockIdx.x;
    const int wave  = threadIdx.x >> 5;               // 0..7 : N tile
    const int lane  = threadIdx.x & 31;
    const int hv    = lane >> 4;                      // K-half selector
    const int l16   = lane & 15;
    const int ncol  = wave * TM + l16;                // B fragment col

    // cooperative strip fill: 256 threads x 16 B == 16x128 bf16
    {
        const __bf16* gsrc = A + (size_t)tileM * TM * DLAT;
        *(v8bf*)(sA + threadIdx.x * 8) = *(const v8bf*)(gsrc + threadIdx.x * 8);
    }
    const __bf16* bp = WbT + (size_t)ncol * DLAT;
    __builtin_prefetch(bp, 0, 0);
    __syncthreads();

    const __bf16* arow = sA + l16 * DLAT;

    v8f c = {};
#pragma unroll
    for (int ks = 0; ks < 4; ++ks) {                  // K = 128
        const int k0 = ks * 32 + hv * 8;
        v8bf alo = *(const v8bf*)(arow + k0);
        v8bf ahi = *(const v8bf*)(arow + k0 + 16);
        v8bf blo = *(const v8bf*)(bp + k0);
        v8bf bhi = *(const v8bf*)(bp + k0 + 16);
        v16bf af  = __builtin_shufflevector(alo, ahi, 0, 1, 2, 3, 4, 5, 6, 7,
                                            8, 9, 10, 11, 12, 13, 14, 15);
        v16bf bfm = __builtin_shufflevector(blo, bhi, 0, 1, 2, 3, 4, 5, 6, 7,
                                            8, 9, 10, 11, 12, 13, 14, 15);
        c = __builtin_amdgcn_wmma_f32_16x16x32_bf16(
                /*neg_a=*/false, af, /*neg_b=*/false, bfm,
                /*c_mod=*/(short)0, c, /*reuse_a=*/false, /*reuse_b=*/false);
    }

    const float bv = bias ? bias[ncol] : 0.0f;
#pragma unroll
    for (int i = 0; i < 8; ++i) {                     // D: M = hv*8 + i, N = lane%16
        float v = c[i] + bv;
        if (ACT == 1) {
            v = 1.0f / (1.0f + __expf(-v));
        } else if (ACT == 2) {
            float t = v - 5.0f;
            v = (t > 20.0f) ? t : log1pf(__expf(t));
        }
        const size_t oidx = (size_t)(tileM * TM + hv * 8 + i) * DLAT + ncol;
        if (OUTBF)
            ((__bf16*)Cout)[oidx] = f2bf(v);
        else
            ((float*)Cout)[oidx] = v;
    }
}

// ---------------------------------------------------------------------------
// Graph aggregation
// ---------------------------------------------------------------------------
__global__ void deg_init(float* __restrict__ deg, int n) {
    int i = blockIdx.x * blockDim.x + threadIdx.x;
    if (i < n) deg[i] = 1.0f;                         // self-loop
}

__global__ void deg_edges(const int* __restrict__ col, float* __restrict__ deg, int E) {
    int e = blockIdx.x * blockDim.x + threadIdx.x;
    if (e < E) atomicAdd(&deg[col[e]], 1.0f);
}

__global__ void deg_rsqrt(float* __restrict__ deg, int n) {
    int i = blockIdx.x * blockDim.x + threadIdx.x;
    if (i < n) {
        float d = deg[i];
        deg[i] = (d > 0.0f) ? rsqrtf(d) : 0.0f;
    }
}

// h = hw * dinv^2 (self-loop) + bias
__global__ void agg_init(const float* __restrict__ hw, const float* __restrict__ dinv,
                         const float* __restrict__ b, float* __restrict__ h, int n) {
    int idx = blockIdx.x * blockDim.x + threadIdx.x;
    if (idx >= n) return;
    int i = idx >> 7, j = idx & 127;
    float di = dinv[i];
    h[idx] = hw[idx] * di * di + b[j];
}

// one wave per edge; 32 lanes x float4 = 128 dims; atomics resolve in L2
__global__ void agg_edges(const int* __restrict__ rowI, const int* __restrict__ colI,
                          const float* __restrict__ hw, const float* __restrict__ dinv,
                          float* __restrict__ h, int E) {
    int e = blockIdx.x * (blockDim.x >> 5) + (threadIdx.x >> 5);
    if (e >= E) return;
    int lane = threadIdx.x & 31;
    int r = rowI[e], cn = colI[e];
    float nrm = dinv[r] * dinv[cn];
    float4 m = ((const float4*)(hw + (size_t)r * DLAT))[lane];
    float* dst = h + (size_t)cn * DLAT + lane * 4;
    atomicAdd(dst + 0, m.x * nrm);
    atomicAdd(dst + 1, m.y * nrm);
    atomicAdd(dst + 2, m.z * nrm);
    atomicAdd(dst + 3, m.w * nrm);
}

// ---------------------------------------------------------------------------
// Reparametrization: z = mu + std * uniform[0,1)  (hash-based RNG)
// ---------------------------------------------------------------------------
__global__ void reparam(const float* __restrict__ mu, const float* __restrict__ sd,
                        float* __restrict__ z, int n) {
    int i = blockIdx.x * blockDim.x + threadIdx.x;
    if (i >= n) return;
    unsigned s = (unsigned)i * 747796405u + 2891336453u;   // PCG-style hash
    unsigned w = ((s >> ((s >> 28) + 4u)) ^ s) * 277803737u;
    w = (w >> 22) ^ w;
    float u = (float)(w >> 8) * (1.0f / 16777216.0f);
    z[i] = mu[i] + sd[i] * u;
}

// ---------------------------------------------------------------------------
extern "C" void kernel_launch(void* const* d_in, const int* in_sizes, int n_in,
                              void* d_out, int out_size, void* d_ws, size_t ws_size,
                              hipStream_t stream) {
    const float* x      = (const float*)d_in[0];
    const int*   ei     = (const int*)d_in[1];
    const float* gcn_w  = (const float*)d_in[2];
    const float* gcn_b  = (const float*)d_in[3];
    const float* enc_w  = (const float*)d_in[4];   // [5][128][128]
    const float* enc_b  = (const float*)d_in[5];   // [5][128]
    const float* mean_w = (const float*)d_in[6];
    const float* mean_b = (const float*)d_in[7];
    const float* std_w  = (const float*)d_in[8];
    const float* std_b  = (const float*)d_in[9];

    const int N  = in_sizes[0] / DLAT;             // 50000
    const int E  = in_sizes[1] / 2;                // 800000
    const int ND = N * DLAT;

    // workspace layout (regions alias across phases; see timeline below)
    char* ws = (char*)d_ws;
    float* dinv = (float*)ws;                                   // N f32
    size_t off = ((size_t)N * 4 + 255) & ~(size_t)255;
    __bf16* WbT = (__bf16*)(ws + off);                          // 8 * 128*128 bf16
    off += ((size_t)8 * DLAT * DLAT * 2 + 255) & ~(size_t)255;
    char* regF = ws + off;                                      // ND f32
    off += (size_t)ND * 4;
    char* regG = ws + off;                                      // ND f32

    // phase views:
    __bf16* xb   = (__bf16*)regF;                 // bf16 copy of x   (phase 1)
    float*  hw   = (float*)regG;                  // x @ W0           (phase 1)
    float*  aggF = (float*)regF;                  // fp32 GCN output  (phase 2; xb dead after gemm0? no!)
    // NOTE: xb must outlive gemm0; hw must outlive agg. Use: xb in regF,
    // hw in regG, agg accumulates into d_out scratch region? Keep it simple:
    // after gemm0 (reads xb/regF, writes hw/regG), agg writes into regF as
    // fp32 (xb is dead then), reading hw/regG. Then cvt regF->bf16 into regG,
    // and ping-pong bf16 layers between regG and regF.
    float*  aggOut = (float*)regF;
    __bf16* hb0    = (__bf16*)regG;               // bf16 h after GCN
    __bf16* hb1    = (__bf16*)regF;

    float* z_out  = (float*)d_out;
    float* mu_out = (float*)d_out + (size_t)ND;
    float* sd_out = (float*)d_out + 2 * (size_t)ND;

    const int WELEM = DLAT * DLAT;
    // 1) weights -> transposed bf16: [0]=gcn, [1..5]=enc, [6]=mean, [7]=std
    cvt_weight_T<<<(WELEM + 255) / 256, 256, 0, stream>>>(gcn_w, WbT);
    for (int i = 0; i < 5; ++i)
        cvt_weight_T<<<(WELEM + 255) / 256, 256, 0, stream>>>(enc_w + (size_t)i * WELEM,
                                                              WbT + (size_t)(1 + i) * WELEM);
    cvt_weight_T<<<(WELEM + 255) / 256, 256, 0, stream>>>(mean_w, WbT + (size_t)6 * WELEM);
    cvt_weight_T<<<(WELEM + 255) / 256, 256, 0, stream>>>(std_w,  WbT + (size_t)7 * WELEM);

    // 2) x -> bf16 (into regF)
    cvt_f32_bf16<<<(ND / 2 + 255) / 256, 256, 0, stream>>>(x, xb, ND / 2);

    // 3) degrees -> dinv
    deg_init<<<(N + 255) / 256, 256, 0, stream>>>(dinv, N);
    deg_edges<<<(E + 255) / 256, 256, 0, stream>>>(ei + E, dinv, E);   // col = edge_index[1]
    deg_rsqrt<<<(N + 255) / 256, 256, 0, stream>>>(dinv, N);

    const int mtiles = N / TM;                     // 3125, exact
    // 4) hw(regG) = xb(regF) @ gcn_w, fp32 out, no bias
    gemm_bf16_wmma<0, 0><<<mtiles, 256, 0, stream>>>(xb, WbT, nullptr, hw);

    // 5) aggregation into regF (xb dead): aggOut = norm-sum(hw) + gcn_b
    agg_init<<<(ND + 255) / 256, 256, 0, stream>>>(hw, dinv, gcn_b, aggOut, ND);
    agg_edges<<<(E + 7) / 8, 256, 0, stream>>>(ei, ei + E, hw, dinv, aggOut, E);

    // 6) fp32 GCN output -> bf16 into regG (hw dead)
    cvt_f32_bf16<<<(ND / 2 + 255) / 256, 256, 0, stream>>>(aggOut, hb0, ND / 2);

    // 7) 5 sigmoid layers, bf16 in / bf16 out, ping-pong regG <-> regF
    __bf16* cur = hb0;
    __bf16* nxt = hb1;
    for (int i = 0; i < 5; ++i) {
        gemm_bf16_wmma<1, 1><<<mtiles, 256, 0, stream>>>(cur, WbT + (size_t)(1 + i) * WELEM,
                                                         enc_b + (size_t)i * DLAT, nxt);
        __bf16* t = cur; cur = nxt; nxt = t;
    }

    // 8) heads: fp32 outputs straight into d_out
    gemm_bf16_wmma<0, 0><<<mtiles, 256, 0, stream>>>(cur, WbT + (size_t)6 * WELEM, mean_b, mu_out);
    gemm_bf16_wmma<2, 0><<<mtiles, 256, 0, stream>>>(cur, WbT + (size_t)7 * WELEM, std_b,  sd_out);

    // 9) z = mu + std * eps
    reparam<<<(ND + 255) / 256, 256, 0, stream>>>(mu_out, sd_out, z_out, ND);
}
